// GraphAttentionEmbedding_21646635172502
// MI455X (gfx1250) — matrix operand order repaired and checked by reading, not compiled, not run on hardware
//
#include <hip/hip_runtime.h>

#define N_NODES 20000
#define E_EDGES 200000
#define IN_CH   256
#define OUT_CH  256
#define HEADS   2
#define CPH     128
#define TDIM_K  100
#define MSGD    156
#define EDGED   256
#define PDROP   0.1f

typedef __attribute__((ext_vector_type(16))) __bf16 v16bf;
typedef __attribute__((ext_vector_type(8)))  float  v8f;

// ---- WMMA operand register layouts (ISA 7.12.2, wave32) -------------------
// A (16x32 bf16): lane = hi*16 + M,  hi = (k5>>3)&1,  idx = (k5&7) + (k5&16 ? 8:0)
// B (32x16 bf16): lane = ((k5>>3)&1)*16 + n15, idx = (k5&7) + (k5&16 ? 8:0)
// C/D (16x16 f32): vgpr r, lane L -> M = r + (L<16?0:8), N = L&15

// float atomic max via int ordering trick (dest initialized to -inf).
__device__ inline void atomic_max_f32(float* addr, float val) {
  if (val >= 0.0f) atomicMax((int*)addr, __float_as_int(val));
  else             atomicMin((unsigned int*)addr, __float_as_uint(val));
}

// -------------------------------------------------------------------------
__global__ void k_init(float* __restrict__ m, float* __restrict__ denom, int n) {
  int i = blockIdx.x * blockDim.x + threadIdx.x;
  if (i < n) { m[i] = -INFINITY; denom[i] = 0.0f; }
}

// -------------------------------------------------------------------------
// Pack x (N x 256 f32) into bf16 WMMA-A tile layout:
// packedX[((rt*8 + kk)*32 + lane)*16 + idx]
__global__ void k_pack_x(const float* __restrict__ x, __bf16* __restrict__ px) {
  int i = blockIdx.x * blockDim.x + threadIdx.x;
  if (i >= N_NODES * IN_CH) return;
  int row = i >> 8, K = i & 255;
  int rt = row >> 4, M = row & 15;
  int kk = K >> 5, k5 = K & 31;
  int hi = (k5 >> 3) & 1;
  int lane = hi * 16 + M;
  int idx = (k5 & 7) + ((k5 & 16) ? 8 : 0);
  px[(((size_t)rt * 8 + kk) * 32 + lane) * 16 + idx] = (__bf16)x[i];
}

// Pack a 256x256 weight matrix into bf16 WMMA-B tile layout:
// packedW[((ct*8 + kk)*32 + lane)*16 + idx]
__global__ void k_pack_w(const float* __restrict__ W, __bf16* __restrict__ pw) {
  int i = blockIdx.x * blockDim.x + threadIdx.x;
  if (i >= IN_CH * OUT_CH) return;
  int K = i >> 8, n = i & 255;
  int ct = n >> 4, n15 = n & 15;
  int kk = K >> 5, k5 = K & 31;
  int lane = (((k5 >> 3) & 1) << 4) | n15;
  int idx = (k5 & 7) + ((k5 & 16) ? 8 : 0);
  pw[(((size_t)ct * 8 + kk) * 32 + lane) * 16 + idx] = (__bf16)W[i];
}

// -------------------------------------------------------------------------
// Node projections q,k,v (-> ws) and skip x@Ws+bs (-> out).
// grid = (N/16, 8), block = 256 (8 waves). wave job in [0,64): proj*16 + coltile.
__global__ __launch_bounds__(256) void k_node_proj(
    const __bf16* __restrict__ px,
    const __bf16* __restrict__ pWq, const float* __restrict__ bq,
    const __bf16* __restrict__ pWk, const float* __restrict__ bk,
    const __bf16* __restrict__ pWv, const float* __restrict__ bv,
    const __bf16* __restrict__ pWs, const float* __restrict__ bs,
    float* __restrict__ q, float* __restrict__ k, float* __restrict__ v,
    float* __restrict__ outskip) {
  const int rt   = blockIdx.x;
  const int wave = threadIdx.x >> 5;
  const int lane = threadIdx.x & 31;
  const int job  = blockIdx.y * 8 + wave;   // 0..63
  const int proj = job >> 4;                // 0..3
  const int ct   = job & 15;                // output col tile

  const __bf16* pW; const float* bias; float* dstp;
  if      (proj == 0) { pW = pWq; bias = bq; dstp = q; }
  else if (proj == 1) { pW = pWk; bias = bk; dstp = k; }
  else if (proj == 2) { pW = pWv; bias = bv; dstp = v; }
  else                { pW = pWs; bias = bs; dstp = outskip; }

  const int M  = lane & 15;
  const int hi = lane >> 4;
  const __bf16* pA = px + (size_t)rt * 4096;   // 8*32*16
  const __bf16* pB = pW + (size_t)ct * 4096;

  v8f acc = {};
#pragma unroll
  for (int kk = 0; kk < IN_CH / 32; ++kk) {
    v16bf a = *(const v16bf*)(pA + (kk * 32 + lane) * 16);
    v16bf b = *(const v16bf*)(pB + (kk * 32 + lane) * 16);
    acc = __builtin_amdgcn_wmma_f32_16x16x32_bf16(false, a, false, b,
                                                  (short)0, acc, false, false);
  }
  const int col = ct * 16 + M;
  const float bb = bias[col];
#pragma unroll
  for (int r = 0; r < 8; ++r) {
    int orow = rt * 16 + r + (hi ? 8 : 0);
    dstp[(size_t)orow * OUT_CH + col] = acc[r] + bb;
  }
}

// -------------------------------------------------------------------------
// Build e = edge_attr @ We + be for one 16-edge tile into LDS eT (16 x 256 f32).
// Step 1 (all 256 threads): build edge_attr tile in bf16 A-layout in LDS aT,
//         cos-time-encoding cols [0,100), msg cols [100,256) -- computed once.
// Step 2 (8 waves x 2 col tiles): WMMA with ds-loaded A, packed global B.
__device__ inline void compute_e_tile(
    int bt, const long long* __restrict__ ei,
    const float* __restrict__ last_update, const float* __restrict__ t,
    const float* __restrict__ msg,
    const float* __restrict__ tw, const float* __restrict__ tb,
    const __bf16* __restrict__ pWe, const float* __restrict__ be,
    v16bf* __restrict__ aT /* LDS [256] */, float* __restrict__ eT /* LDS [16*256] */) {
  const int row = threadIdx.x >> 4;        // edge row 0..15
  const int sub = threadIdx.x & 15;        // 16-wide K chunk
  const int edge = bt * 16 + row;
  const int src  = (int)ei[edge];
  const float rel_t = last_update[src] - t[edge];
  __bf16* aw = (__bf16*)aT;
#pragma unroll
  for (int i = 0; i < 16; ++i) {
    int K = sub * 16 + i;
    float val = (K < TDIM_K) ? __builtin_cosf(rel_t * tw[K] + tb[K])
                             : msg[(size_t)edge * MSGD + (K - TDIM_K)];
    int kk = K >> 5, k5 = K & 31;
    int hi = (k5 >> 3) & 1;
    int lane2 = hi * 16 + row;
    int idx = (k5 & 7) + ((k5 & 16) ? 8 : 0);
    aw[(kk * 32 + lane2) * 16 + idx] = (__bf16)val;
  }
  __syncthreads();

  const int wave = threadIdx.x >> 5;
  const int lane = threadIdx.x & 31;
  const int M  = lane & 15;
  const int hi = lane >> 4;
#pragma unroll
  for (int c = 0; c < 2; ++c) {
    const int ct  = wave * 2 + c;
    const int col = ct * 16 + M;
    const __bf16* pB = pWe + (size_t)ct * 4096;
    v8f acc = {};
#pragma unroll
    for (int kk = 0; kk < EDGED / 32; ++kk) {
      v16bf a = aT[kk * 32 + lane];
      v16bf b = *(const v16bf*)(pB + (kk * 32 + lane) * 16);
      acc = __builtin_amdgcn_wmma_f32_16x16x32_bf16(false, a, false, b,
                                                    (short)0, acc, false, false);
    }
    const float bb = be[col];
#pragma unroll
    for (int r = 0; r < 8; ++r)
      eT[(r + (hi ? 8 : 0)) * EDGED + col] = acc[r] + bb;
  }
}

// -------------------------------------------------------------------------
// Pass A: alpha[e][h] = q[dst].(k[src]+e)/sqrt(C), edge-dropout mask,
// atomic segment max into m. grid = E/16, block = 256.
__global__ __launch_bounds__(256) void k_edge_alpha(
    const long long* __restrict__ ei,
    const float* __restrict__ last_update, const float* __restrict__ t,
    const float* __restrict__ msg, const float* __restrict__ u_edge,
    const float* __restrict__ tw, const float* __restrict__ tb,
    const __bf16* __restrict__ pWe, const float* __restrict__ be,
    const float* __restrict__ q, const float* __restrict__ k,
    float* __restrict__ alpha, float* __restrict__ m) {
  __shared__ v16bf aT[256];
  __shared__ float eT[16 * EDGED];
  __shared__ float part[16][16];
  const int bt = blockIdx.x;
  compute_e_tile(bt, ei, last_update, t, msg, tw, tb, pWe, be, aT, eT);
  __syncthreads();

  const int e_i  = threadIdx.x >> 4;
  const int sub  = threadIdx.x & 15;
  const int edge = bt * 16 + e_i;
  const int src  = (int)ei[edge];
  const int dn   = (int)ei[E_EDGES + edge];
  float s = 0.0f;
  const int c0 = sub * 16;
#pragma unroll
  for (int j = 0; j < 16; ++j) {
    int c = c0 + j;
    s += q[(size_t)dn * OUT_CH + c] *
         (k[(size_t)src * OUT_CH + c] + eT[e_i * EDGED + c]);
  }
  part[e_i][sub] = s;
  __syncthreads();

  if (threadIdx.x < 32) {
    int e2 = threadIdx.x >> 1;
    int h  = threadIdx.x & 1;
    int edge2 = bt * 16 + e2;
    float a = 0.0f;
#pragma unroll
    for (int j = 0; j < 8; ++j) a += part[e2][h * 8 + j];
    a *= 0.08838834764831845f;             // 1/sqrt(128)
    if (!(u_edge[edge2] > PDROP)) a = -INFINITY;
    int dn2 = (int)ei[E_EDGES + edge2];
    alpha[edge2 * HEADS + h] = a;
    atomic_max_f32(&m[dn2 * HEADS + h], a);
  }
}

// -------------------------------------------------------------------------
__global__ void k_denom(const long long* __restrict__ ei,
                        const float* __restrict__ alpha,
                        const float* __restrict__ m,
                        float* __restrict__ denom) {
  int i = blockIdx.x * blockDim.x + threadIdx.x;
  if (i >= E_EDGES * HEADS) return;
  int edge = i >> 1, h = i & 1;
  int dn = (int)ei[E_EDGES + edge];
  float mm = m[dn * HEADS + h];
  mm = (mm == -INFINITY) ? 0.0f : mm;
  float ex = __expf(alpha[i] - mm);        // 0 for dropped (-inf) edges
  if (ex != 0.0f) atomicAdd(&denom[dn * HEADS + h], ex);
}

// -------------------------------------------------------------------------
// Pass C: recompute e (WMMA), attn = softmax * attn-dropout, scatter-add
// (v[src]+e)*attn into out. grid = E/16, block = 256.
__global__ __launch_bounds__(256) void k_edge_scatter(
    const long long* __restrict__ ei,
    const float* __restrict__ last_update, const float* __restrict__ t,
    const float* __restrict__ msg, const float* __restrict__ u_attn,
    const float* __restrict__ tw, const float* __restrict__ tb,
    const __bf16* __restrict__ pWe, const float* __restrict__ be,
    const float* __restrict__ v, const float* __restrict__ alpha,
    const float* __restrict__ m, const float* __restrict__ denom,
    float* __restrict__ out) {
  __shared__ v16bf aT[256];
  __shared__ float eT[16 * EDGED];
  const int bt = blockIdx.x;
  compute_e_tile(bt, ei, last_update, t, msg, tw, tb, pWe, be, aT, eT);
  __syncthreads();

  const int e_i  = threadIdx.x >> 4;
  const int sub  = threadIdx.x & 15;
  const int h    = sub >> 3;               // cols 0..127 -> h0, 128..255 -> h1
  const int edge = bt * 16 + e_i;
  const int src  = (int)ei[edge];
  const int dn   = (int)ei[E_EDGES + edge];

  float mm = m[dn * HEADS + h];
  mm = (mm == -INFINITY) ? 0.0f : mm;
  float ex   = __expf(alpha[edge * HEADS + h] - mm);
  float attn = ex / (denom[dn * HEADS + h] + 1e-16f);
  attn *= (u_attn[edge * HEADS + h] > PDROP) ? (1.0f / (1.0f - PDROP)) : 0.0f;
  if (attn != 0.0f) {
    const int c0 = sub * 16;
#pragma unroll
    for (int j = 0; j < 16; ++j) {
      int c = c0 + j;
      float val = (v[(size_t)src * OUT_CH + c] + eT[e_i * EDGED + c]) * attn;
      atomicAdd(&out[(size_t)dn * OUT_CH + c], val);
    }
  }
}

// -------------------------------------------------------------------------
extern "C" void kernel_launch(void* const* d_in, const int* in_sizes, int n_in,
                              void* d_out, int out_size, void* d_ws, size_t ws_size,
                              hipStream_t stream) {
  const float*     x           = (const float*)d_in[0];
  const float*     last_update = (const float*)d_in[1];
  const long long* ei          = (const long long*)d_in[2];   // int64 [2,E]
  const float*     t           = (const float*)d_in[3];
  const float*     msg         = (const float*)d_in[4];
  const float*     u_edge      = (const float*)d_in[5];
  const float*     u_attn      = (const float*)d_in[6];
  const float*     tw          = (const float*)d_in[7];
  const float*     tb          = (const float*)d_in[8];
  const float* Wq = (const float*)d_in[9];  const float* bq = (const float*)d_in[10];
  const float* Wk = (const float*)d_in[11]; const float* bk = (const float*)d_in[12];
  const float* Wv = (const float*)d_in[13]; const float* bv = (const float*)d_in[14];
  const float* We = (const float*)d_in[15]; const float* be = (const float*)d_in[16];
  const float* Ws = (const float*)d_in[17]; const float* bs = (const float*)d_in[18];
  float* out = (float*)d_out;

  // workspace layout (~74.3 MB total); all offsets 32B-aligned.
  char* ws = (char*)d_ws;
  size_t off = 0;
  float* q     = (float*)(ws + off); off += (size_t)N_NODES * OUT_CH * sizeof(float);
  float* k     = (float*)(ws + off); off += (size_t)N_NODES * OUT_CH * sizeof(float);
  float* v     = (float*)(ws + off); off += (size_t)N_NODES * OUT_CH * sizeof(float);
  float* alpha = (float*)(ws + off); off += (size_t)E_EDGES * HEADS * sizeof(float);
  float* m     = (float*)(ws + off); off += (size_t)N_NODES * HEADS * sizeof(float);
  float* denom = (float*)(ws + off); off += (size_t)N_NODES * HEADS * sizeof(float);
  __bf16* px   = (__bf16*)(ws + off); off += (size_t)N_NODES * IN_CH * sizeof(__bf16);
  __bf16* pWq  = (__bf16*)(ws + off); off += (size_t)IN_CH * OUT_CH * sizeof(__bf16);
  __bf16* pWk  = (__bf16*)(ws + off); off += (size_t)IN_CH * OUT_CH * sizeof(__bf16);
  __bf16* pWv  = (__bf16*)(ws + off); off += (size_t)IN_CH * OUT_CH * sizeof(__bf16);
  __bf16* pWs  = (__bf16*)(ws + off); off += (size_t)IN_CH * OUT_CH * sizeof(__bf16);
  __bf16* pWe  = (__bf16*)(ws + off); off += (size_t)EDGED * OUT_CH * sizeof(__bf16);

  k_init<<<(N_NODES * HEADS + 255) / 256, 256, 0, stream>>>(m, denom, N_NODES * HEADS);
  k_pack_x<<<(N_NODES * IN_CH + 255) / 256, 256, 0, stream>>>(x, px);
  k_pack_w<<<(IN_CH * OUT_CH + 255) / 256, 256, 0, stream>>>(Wq, pWq);
  k_pack_w<<<(IN_CH * OUT_CH + 255) / 256, 256, 0, stream>>>(Wk, pWk);
  k_pack_w<<<(IN_CH * OUT_CH + 255) / 256, 256, 0, stream>>>(Wv, pWv);
  k_pack_w<<<(IN_CH * OUT_CH + 255) / 256, 256, 0, stream>>>(Ws, pWs);
  k_pack_w<<<(EDGED * OUT_CH + 255) / 256, 256, 0, stream>>>(We, pWe);

  k_node_proj<<<dim3(N_NODES / 16, 8), 256, 0, stream>>>(
      px, pWq, bq, pWk, bk, pWv, bv, pWs, bs, q, k, v, out);
  k_edge_alpha<<<E_EDGES / 16, 256, 0, stream>>>(
      ei, last_update, t, msg, u_edge, tw, tb, pWe, be, q, k, alpha, m);
  k_denom<<<(E_EDGES * HEADS + 255) / 256, 256, 0, stream>>>(ei, alpha, m, denom);
  k_edge_scatter<<<E_EDGES / 16, 256, 0, stream>>>(
      ei, last_update, t, msg, u_attn, tw, tb, pWe, be, v, alpha, m, denom, out);
}
